// MultiHeadSelfAttention_30270929502667
// MI455X (gfx1250) — matrix operand order repaired and checked by reading, not compiled, not run on hardware
//
#include <hip/hip_runtime.h>

// ---- problem constants ----
#define EDIM 1024
#define SEQ  2048
#define NBAT 4
#define NHEAD 16
#define HDIM 64
#define MTOT (NBAT * SEQ)   // 8192

typedef __attribute__((ext_vector_type(16))) __bf16 v16bf;
typedef __attribute__((ext_vector_type(8)))  float  v8f;
typedef int v4i __attribute__((vector_size(16)));   // matches builtin param type

union Frag {
    v16bf v;
    uint4 q[2];
    unsigned short u[16];
};
union Pack8 {
    unsigned short s[8];
    uint4 q;
};

__device__ __forceinline__ unsigned short f2bf(float f) {
    union { float f; unsigned int u; } a;
    a.f = f;
    unsigned int u = a.u;
    u += 0x7fffu + ((u >> 16) & 1u);   // round-to-nearest-even
    return (unsigned short)(u >> 16);
}

// A fragment (16x32 bf16, MxK): lane holds row M=lane%16.
// u[0..7] = K[8g..8g+7], u[8..15] = K[16+8g..23+8g], g = lane/16.
__device__ __forceinline__ void load_frag_a(Frag& f, const unsigned short* row, int g) {
    const uint4* p = (const uint4*)row;
    f.q[0] = p[g];
    f.q[1] = p[2 + g];
}

// B fragment (32x16 bf16, KxN): lane holds col N=lane%16.
// u[0..7] = K[8g..8g+7], u[8..15] = K[16+8g..23+8g]; row = 32 contiguous bf16 of that column's k-run.
__device__ __forceinline__ void load_frag_b(Frag& f, const unsigned short* row, int g) {
    const uint4* p = (const uint4*)row;
    f.q[0] = p[g];
    f.q[1] = p[2 + g];
}

__device__ __forceinline__ v8f wmma_bf16(const Frag& a, const Frag& b, v8f c) {
    return __builtin_amdgcn_wmma_f32_16x16x32_bf16(
        false, a.v, false, b.v, (short)0, c, false, false);
}

__device__ __forceinline__ v8f zero_v8f() {
    v8f z;
#pragma unroll
    for (int i = 0; i < 8; ++i) z[i] = 0.0f;
    return z;
}

// ---- async global->LDS copy (CDNA5), with safe fallback ----
__device__ __forceinline__ void async_copy16(void* lds_dst, const void* gsrc) {
#if defined(__gfx1250__) && __has_builtin(__builtin_amdgcn_global_load_async_to_lds_b128)
    __builtin_amdgcn_global_load_async_to_lds_b128(
        (__attribute__((address_space(1))) v4i*)gsrc,
        (__attribute__((address_space(3))) v4i*)lds_dst,
        0, 0);
#else
    *(uint4*)lds_dst = *(const uint4*)gsrc;
#endif
}
__device__ __forceinline__ void async_wait0() {
#if defined(__gfx1250__) && __has_builtin(__builtin_amdgcn_global_load_async_to_lds_b128)
#if __has_builtin(__builtin_amdgcn_s_wait_asynccnt)
    __builtin_amdgcn_s_wait_asynccnt(0);
#else
    asm volatile("s_wait_asynccnt 0x0" ::: "memory");
#endif
#endif
}

// =====================================================================
// Prep A: x fp32 -> bf16 (same row-major layout), one pass
// =====================================================================
__global__ __launch_bounds__(256)
void convert_x_kernel(const float* __restrict__ x, unsigned short* __restrict__ xb) {
    const size_t i = ((size_t)blockIdx.x * 256 + threadIdx.x) * 8;
    float4 f0 = *(const float4*)(x + i);
    float4 f1 = *(const float4*)(x + i + 4);
    Pack8 p;
    p.s[0] = f2bf(f0.x); p.s[1] = f2bf(f0.y); p.s[2] = f2bf(f0.z); p.s[3] = f2bf(f0.w);
    p.s[4] = f2bf(f1.x); p.s[5] = f2bf(f1.y); p.s[6] = f2bf(f1.z); p.s[7] = f2bf(f1.w);
    *(uint4*)(xb + i) = p.q;
}

// =====================================================================
// Prep B: W[k][n] fp32 -> Wt[n][k] bf16 for the 4 weight matrices
// =====================================================================
__global__ __launch_bounds__(256)
void transpose_w_kernel(const float* __restrict__ Wq, const float* __restrict__ Wk,
                        const float* __restrict__ Wv, const float* __restrict__ Wo,
                        unsigned short* __restrict__ wt)
{
    __shared__ unsigned short lt[64][65];   // padded vs bank conflicts
    const int z = blockIdx.z;
    const float* W = (z == 0) ? Wq : (z == 1) ? Wk : (z == 2) ? Wv : Wo;
    unsigned short* Wt = wt + (size_t)z * EDIM * EDIM;
    const int k0 = blockIdx.x * 64;
    const int n0 = blockIdx.y * 64;
    const int t  = threadIdx.x;
    {
        const int i  = t >> 2;           // k within tile
        const int j0 = (t & 3) * 16;     // n within tile
        const float* src = W + (size_t)(k0 + i) * EDIM + n0 + j0;
#pragma unroll
        for (int j = 0; j < 16; j += 4) {
            float4 f = *(const float4*)(src + j);
            lt[i][j0 + j + 0] = f2bf(f.x);
            lt[i][j0 + j + 1] = f2bf(f.y);
            lt[i][j0 + j + 2] = f2bf(f.z);
            lt[i][j0 + j + 3] = f2bf(f.w);
        }
    }
    __syncthreads();
    {
        const int n  = t >> 2;
        const int kb = (t & 3) * 16;
        Pack8 a, b;
#pragma unroll
        for (int j = 0; j < 8; ++j) a.s[j] = lt[kb + j][n];
#pragma unroll
        for (int j = 0; j < 8; ++j) b.s[j] = lt[kb + 8 + j][n];
        uint4* dst = (uint4*)(Wt + (size_t)(n0 + n) * EDIM + k0 + kb);
        dst[0] = a.q;
        dst[1] = b.q;
    }
}

// =====================================================================
// Kernel 1: QKV GEMM, pure register/WMMA (no LDS, no barriers).
// block=256 (8 waves), each wave: 16 rows x 64 cols, k-step 64.
// Q,K -> [B*H, S, 64] bf16 ; V -> transposed [B*H, 64, S] bf16
// =====================================================================
__global__ __launch_bounds__(256)
void qkv_gemm_kernel(const unsigned short* __restrict__ xb,
                     const unsigned short* __restrict__ wt,
                     const float* __restrict__ bq, const float* __restrict__ bk,
                     const float* __restrict__ bv,
                     unsigned short* __restrict__ qo,
                     unsigned short* __restrict__ ko,
                     unsigned short* __restrict__ vto)
{
    const int which = blockIdx.z;            // 0=Q 1=K 2=V
    const unsigned short* Wt = wt + (size_t)which * EDIM * EDIM;
    const float* bias = (which == 0) ? bq : (which == 1) ? bk : bv;

    const int m0 = blockIdx.x * 128;
    const int n0 = blockIdx.y * 64;
    const int tid  = threadIdx.x;
    const int lane = tid & 31;
    const int wv   = tid >> 5;
    const int g    = (lane >> 4) & 1;
    const int ln16 = lane & 15;

    v8f acc[4];
#pragma unroll
    for (int t = 0; t < 4; ++t) acc[t] = zero_v8f();

    const unsigned short* arow = xb + (size_t)(m0 + wv * 16 + ln16) * EDIM;
    const unsigned short* brow0 = Wt + (size_t)(n0 +  0 + ln16) * EDIM;
    const unsigned short* brow1 = Wt + (size_t)(n0 + 16 + ln16) * EDIM;
    const unsigned short* brow2 = Wt + (size_t)(n0 + 32 + ln16) * EDIM;
    const unsigned short* brow3 = Wt + (size_t)(n0 + 48 + ln16) * EDIM;

    for (int kk = 0; kk < EDIM; kk += 64) {
        Frag a0, a1, b0, b1;
        load_frag_a(a0, arow + kk, g);
        load_frag_a(a1, arow + kk + 32, g);

        load_frag_b(b0, brow0 + kk, g);
        load_frag_b(b1, brow0 + kk + 32, g);
        acc[0] = wmma_bf16(a0, b0, acc[0]);
        acc[0] = wmma_bf16(a1, b1, acc[0]);

        load_frag_b(b0, brow1 + kk, g);
        load_frag_b(b1, brow1 + kk + 32, g);
        acc[1] = wmma_bf16(a0, b0, acc[1]);
        acc[1] = wmma_bf16(a1, b1, acc[1]);

        load_frag_b(b0, brow2 + kk, g);
        load_frag_b(b1, brow2 + kk + 32, g);
        acc[2] = wmma_bf16(a0, b0, acc[2]);
        acc[2] = wmma_bf16(a1, b1, acc[2]);

        load_frag_b(b0, brow3 + kk, g);
        load_frag_b(b1, brow3 + kk + 32, g);
        acc[3] = wmma_bf16(a0, b0, acc[3]);
        acc[3] = wmma_bf16(a1, b1, acc[3]);
    }

#pragma unroll
    for (int nt = 0; nt < 4; ++nt) {
        const int n = n0 + nt * 16 + ln16;
        const float bval = bias[n];
        const int h = n >> 6;
        const int d = n & 63;
#pragma unroll
        for (int v = 0; v < 8; ++v) {
            const int m  = m0 + wv * 16 + v + 8 * g;
            const int b_ = m >> 11;
            const int s  = m & (SEQ - 1);
            const size_t bh = (size_t)b_ * NHEAD + h;
            const unsigned short val = f2bf(acc[nt][v] + bval);
            if (which == 2) {
                vto[(bh * HDIM + d) * SEQ + s] = val;
            } else {
                unsigned short* o = which ? ko : qo;
                o[(bh * SEQ + s) * HDIM + d] = val;
            }
        }
    }
}

// =====================================================================
// Kernel 2: flash attention. block=256 (8 waves, 128 queries).
// K/V tiles staged once per block in LDS (async-to-LDS when available),
// shared by all 8 waves. Online softmax in f32.
// =====================================================================
__global__ __launch_bounds__(256)
void flash_attn_kernel(const unsigned short* __restrict__ q,
                       const unsigned short* __restrict__ k,
                       const unsigned short* __restrict__ vt,
                       unsigned short* __restrict__ attn)
{
    __shared__ unsigned short lk[64][64];      // K tile  [key][d]
    __shared__ unsigned short lv[64][64];      // Vt tile [d][key]
    __shared__ unsigned short lp[8][16][64];   // per-wave P staging

    const int bh  = blockIdx.y;
    const int q0  = blockIdx.x * 128;
    const int tid  = threadIdx.x;
    const int lane = tid & 31;
    const int wv   = tid >> 5;                 // wave 0..7
    const int g    = (lane >> 4) & 1;
    const int ln16 = lane & 15;
    const int b_ = bh >> 4;
    const int h  = bh & 15;

    const unsigned short* qrow = q + ((size_t)bh * SEQ + q0 + wv * 16 + ln16) * HDIM;
    Frag qf0, qf1;
    load_frag_a(qf0, qrow, g);
    load_frag_a(qf1, qrow + 32, g);

    v8f oacc[4];
#pragma unroll
    for (int t = 0; t < 4; ++t) oacc[t] = zero_v8f();
    float mrow[8], lrow[8];
#pragma unroll
    for (int v = 0; v < 8; ++v) { mrow[v] = -__builtin_inff(); lrow[v] = 0.0f; }

    const int sr = tid >> 2;          // staging row 0..63
    const int sc = (tid & 3) * 16;    // staging col 0,16,32,48

    for (int kb = 0; kb < SEQ; kb += 64) {
        // ---- stage K (64x64) and Vt (64x64) tiles into LDS ----
        {
            const unsigned short* ksrc = k  + ((size_t)bh * SEQ  + kb + sr) * HDIM;
            const unsigned short* vsrc = vt + ((size_t)bh * HDIM + sr) * SEQ + kb;
            async_copy16(&lk[sr][sc],     ksrc + sc);
            async_copy16(&lk[sr][sc + 8], ksrc + sc + 8);
            async_copy16(&lv[sr][sc],     vsrc + sc);
            async_copy16(&lv[sr][sc + 8], vsrc + sc + 8);
        }
        async_wait0();
        __syncthreads();

        // ---- S = Q @ K^T (16 x 64 per wave), scale 1/sqrt(64) ----
        v8f scv[4];
#pragma unroll
        for (int nt = 0; nt < 4; ++nt) {
            const unsigned short* krow = &lk[nt * 16 + ln16][0];
            Frag kf0, kf1;
            load_frag_b(kf0, krow, g);
            load_frag_b(kf1, krow + 32, g);
            v8f t = zero_v8f();
            t = wmma_bf16(qf0, kf0, t);
            t = wmma_bf16(qf1, kf1, t);
            scv[nt] = t;
        }

        // ---- online softmax; rows m = v + 8g, cols across 16 lanes ----
        float alpha[8];
#pragma unroll
        for (int v = 0; v < 8; ++v) {
            float mx = -__builtin_inff();
#pragma unroll
            for (int nt = 0; nt < 4; ++nt) {
                scv[nt][v] *= 0.125f;
                mx = fmaxf(mx, scv[nt][v]);
            }
#pragma unroll
            for (int off = 1; off < 16; off <<= 1)
                mx = fmaxf(mx, __shfl_xor(mx, off, 32));
            const float mnew = fmaxf(mrow[v], mx);
            alpha[v] = __expf(mrow[v] - mnew);
            float rs = 0.0f;
#pragma unroll
            for (int nt = 0; nt < 4; ++nt) {
                const float p = __expf(scv[nt][v] - mnew);
                scv[nt][v] = p;
                rs += p;
            }
#pragma unroll
            for (int off = 1; off < 16; off <<= 1)
                rs += __shfl_xor(rs, off, 32);
            lrow[v] = lrow[v] * alpha[v] + rs;
            mrow[v] = mnew;
        }

        // ---- P: D layout -> A layout through per-wave LDS ----
#pragma unroll
        for (int nt = 0; nt < 4; ++nt)
#pragma unroll
            for (int v = 0; v < 8; ++v)
                lp[wv][v + 8 * g][nt * 16 + ln16] = f2bf(scv[nt][v]);
        asm volatile("s_wait_dscnt 0" ::: "memory");

        Frag pf0, pf1;
        load_frag_a(pf0, &lp[wv][ln16][0], g);
        load_frag_a(pf1, &lp[wv][ln16][32], g);

        // ---- O = O*alpha + P @ V ----
#pragma unroll
        for (int dt = 0; dt < 4; ++dt) {
#pragma unroll
            for (int v = 0; v < 8; ++v) oacc[dt][v] *= alpha[v];
            const unsigned short* vrow = &lv[dt * 16 + ln16][0];
            Frag vf0, vf1;
            load_frag_b(vf0, vrow, g);
            load_frag_b(vf1, vrow + 32, g);
            oacc[dt] = wmma_bf16(pf0, vf0, oacc[dt]);
            oacc[dt] = wmma_bf16(pf1, vf1, oacc[dt]);
        }
        __syncthreads();   // all frag reads done before next tile overwrites LDS
    }

    float inv[8];
#pragma unroll
    for (int v = 0; v < 8; ++v) inv[v] = 1.0f / lrow[v];
#pragma unroll
    for (int dt = 0; dt < 4; ++dt) {
        const int d = dt * 16 + ln16;
#pragma unroll
        for (int v = 0; v < 8; ++v) {
            const int s = q0 + wv * 16 + v + 8 * g;
            attn[((size_t)b_ * SEQ + s) * EDIM + h * HDIM + d] =
                f2bf(oacc[dt][v] * inv[v]);
        }
    }
}

// =====================================================================
// Kernel 3: output projection (attn_bf16 @ Wo_t + bo -> fp32), no LDS.
// =====================================================================
__global__ __launch_bounds__(256)
void out_gemm_kernel(const unsigned short* __restrict__ attn,
                     const unsigned short* __restrict__ wto,
                     const float* __restrict__ bo,
                     float* __restrict__ out)
{
    const int m0 = blockIdx.x * 128;
    const int n0 = blockIdx.y * 64;
    const int tid  = threadIdx.x;
    const int lane = tid & 31;
    const int wv   = tid >> 5;
    const int g    = (lane >> 4) & 1;
    const int ln16 = lane & 15;

    v8f acc[4];
#pragma unroll
    for (int t = 0; t < 4; ++t) acc[t] = zero_v8f();

    const unsigned short* arow = attn + (size_t)(m0 + wv * 16 + ln16) * EDIM;
    const unsigned short* brow0 = wto + (size_t)(n0 +  0 + ln16) * EDIM;
    const unsigned short* brow1 = wto + (size_t)(n0 + 16 + ln16) * EDIM;
    const unsigned short* brow2 = wto + (size_t)(n0 + 32 + ln16) * EDIM;
    const unsigned short* brow3 = wto + (size_t)(n0 + 48 + ln16) * EDIM;

    for (int kk = 0; kk < EDIM; kk += 64) {
        Frag a0, a1, b0, b1;
        load_frag_a(a0, arow + kk, g);
        load_frag_a(a1, arow + kk + 32, g);

        load_frag_b(b0, brow0 + kk, g);
        load_frag_b(b1, brow0 + kk + 32, g);
        acc[0] = wmma_bf16(a0, b0, acc[0]);
        acc[0] = wmma_bf16(a1, b1, acc[0]);

        load_frag_b(b0, brow1 + kk, g);
        load_frag_b(b1, brow1 + kk + 32, g);
        acc[1] = wmma_bf16(a0, b0, acc[1]);
        acc[1] = wmma_bf16(a1, b1, acc[1]);

        load_frag_b(b0, brow2 + kk, g);
        load_frag_b(b1, brow2 + kk + 32, g);
        acc[2] = wmma_bf16(a0, b0, acc[2]);
        acc[2] = wmma_bf16(a1, b1, acc[2]);

        load_frag_b(b0, brow3 + kk, g);
        load_frag_b(b1, brow3 + kk + 32, g);
        acc[3] = wmma_bf16(a0, b0, acc[3]);
        acc[3] = wmma_bf16(a1, b1, acc[3]);
    }

#pragma unroll
    for (int nt = 0; nt < 4; ++nt) {
        const int n = n0 + nt * 16 + ln16;
        const float bval = bo[n];
#pragma unroll
        for (int v = 0; v < 8; ++v) {
            const int m = m0 + wv * 16 + v + 8 * g;
            out[(size_t)m * EDIM + n] = acc[nt][v] + bval;
        }
    }
}

// =====================================================================
extern "C" void kernel_launch(void* const* d_in, const int* in_sizes, int n_in,
                              void* d_out, int out_size, void* d_ws, size_t ws_size,
                              hipStream_t stream) {
    (void)in_sizes; (void)n_in; (void)out_size; (void)ws_size;

    const float* x  = (const float*)d_in[0];
    const float* Wq = (const float*)d_in[1];
    const float* bq = (const float*)d_in[2];
    const float* Wk = (const float*)d_in[3];
    const float* bk = (const float*)d_in[4];
    const float* Wv = (const float*)d_in[5];
    const float* bv = (const float*)d_in[6];
    const float* Wo = (const float*)d_in[7];
    const float* bo = (const float*)d_in[8];
    float* out = (float*)d_out;

    const size_t XE = (size_t)MTOT * EDIM;          // 8M
    const size_t WE = (size_t)EDIM * EDIM;          // 1M
    unsigned short* xb   = (unsigned short*)d_ws;   // 8M
    unsigned short* wt   = xb + XE;                 // 4M (Q,K,V,O transposed)
    unsigned short* qws  = wt + 4 * WE;             // 8M
    unsigned short* kws  = qws + XE;                // 8M
    unsigned short* vtws = kws + XE;                // 8M
    unsigned short* aws  = vtws + XE;               // 8M

    convert_x_kernel<<<(int)(XE / (256 * 8)), 256, 0, stream>>>(x, xb);

    dim3 gT(EDIM / 64, EDIM / 64, 4);               // 16 x 16 x 4
    transpose_w_kernel<<<gT, 256, 0, stream>>>(Wq, Wk, Wv, Wo, wt);

    dim3 g1(MTOT / 128, EDIM / 64, 3);              // 64 x 16 x 3
    qkv_gemm_kernel<<<g1, 256, 0, stream>>>(xb, wt, bq, bk, bv, qws, kws, vtws);

    dim3 g2(SEQ / 128, NBAT * NHEAD);               // 16 x 64
    flash_attn_kernel<<<g2, 256, 0, stream>>>(qws, kws, vtws, aws);

    dim3 g3(MTOT / 128, EDIM / 64);                 // 64 x 16
    out_gemm_kernel<<<g3, 256, 0, stream>>>(aws, wt + 3 * WE, bo, out);
}
